// IcoBlock_65678639891284
// MI455X (gfx1250) — matrix operand order repaired and checked by reading, not compiled, not run on hardware
//
#include <hip/hip_runtime.h>
#include <hip/hip_bf16.h>
#include <math.h>

// ---------------------------------------------------------------- constants
#define NVERT 2562
#define DEPTH 8
#define CDIM  512
#define HEADS 8
#define HD    64
#define KNBR  8
#define HID   2048
#define MROWS (NVERT * DEPTH)   // 20496 tokens
#define NQKV  (3 * CDIM)        // 1536

typedef __attribute__((ext_vector_type(16))) __bf16 v16bf;
typedef __attribute__((ext_vector_type(8)))  __bf16 v8bf;
typedef __attribute__((ext_vector_type(8)))  float  v8f;

// ------------------------------------------------------- bf16 <-> f32 bits
static __device__ __forceinline__ float bf2f(__bf16 v) {
    unsigned short u = __builtin_bit_cast(unsigned short, v);
    unsigned int   x = ((unsigned int)u) << 16;
    return __builtin_bit_cast(float, x);
}
static __device__ __forceinline__ __bf16 f2bf(float f) {
    unsigned int x = __builtin_bit_cast(unsigned int, f);
    unsigned int lsb = (x >> 16) & 1u;
    x += 0x7fffu + lsb;                     // round-to-nearest-even
    unsigned short u = (unsigned short)(x >> 16);
    return __builtin_bit_cast(__bf16, u);
}

// --------------------------------------------------------- fragment loads
// A fragment (16x32, MxK): lane<16 -> K {0..7,16..23}; lane>=16 -> +8.
// p must already include row*ldk + k0 + kh  (kh = (lane>>4)*8)
static __device__ __forceinline__ v16bf load_a_frag(const __bf16* p) {
    v8bf lo = *(const v8bf*)(p);
    v8bf hi = *(const v8bf*)(p + 16);
    v16bf r;
#pragma unroll
    for (int i = 0; i < 8; ++i) { r[i] = lo[i]; r[i + 8] = hi[i]; }
    return r;
}
// B fragment (32x16, KxN): col = lane&15; lane<16 -> K {0..7,16..23}; lane>=16 -> +8.
// p must already include col*ldk + k0 + (lane>>4)*8
static __device__ __forceinline__ v16bf load_b_frag(const __bf16* p) {
    v8bf lo = *(const v8bf*)(p);
    v8bf hi = *(const v8bf*)(p + 16);
    v16bf r;
#pragma unroll
    for (int i = 0; i < 8; ++i) { r[i] = lo[i]; r[i + 8] = hi[i]; }
    return r;
}

static __device__ __forceinline__ v8f wmma_bf16(v16bf a, v16bf b, v8f c) {
    return __builtin_amdgcn_wmma_f32_16x16x32_bf16(false, a, false, b,
                                                   (short)0, c, false, false);
}

// ------------------------------------------------------------ weight cast
__global__ void cast_f32_to_bf16(const float* __restrict__ in,
                                 __bf16* __restrict__ out, int n) {
    for (int i = blockIdx.x * blockDim.x + threadIdx.x; i < n;
         i += gridDim.x * blockDim.x)
        out[i] = f2bf(in[i]);
}

// --------------------------------------------------------- layernorm->bf16
__global__ void layernorm_bf16(const float* __restrict__ x,
                               const float* __restrict__ w,
                               const float* __restrict__ b,
                               __bf16* __restrict__ out) {
    const int row = blockIdx.x;
    const int t   = threadIdx.x;                 // 256 threads, C=512
    const float* xr = x + (size_t)row * CDIM;
    float a0 = xr[t], a1 = xr[t + 256];
    __shared__ float s1[256], s2[256];
    s1[t] = a0 + a1;
    s2[t] = a0 * a0 + a1 * a1;
    __syncthreads();
    for (int off = 128; off > 0; off >>= 1) {
        if (t < off) { s1[t] += s1[t + off]; s2[t] += s2[t + off]; }
        __syncthreads();
    }
    const float mean = s1[0] * (1.0f / CDIM);
    const float var  = s2[0] * (1.0f / CDIM) - mean * mean;
    const float rinv = rsqrtf(var + 1e-5f);
    __bf16* orow = out + (size_t)row * CDIM;
    orow[t]       = f2bf((a0 - mean) * rinv * w[t]       + b[t]);
    orow[t + 256] = f2bf((a1 - mean) * rinv * w[t + 256] + b[t + 256]);
}

// --------------------------------------------------------------- WMMA GEMM
// out[M,N] = A[M,K] @ W[N,K]^T (+bias) (+epilogue)
// Each wave computes a 32x64 output tile: 2 A-frags x 4 B-frags = 8 WMMA
// per K-step of 32 against 12 b128 loads (~10.9 MAC/byte from L2).
// MODE 0: bf16 out = acc + bias                     (QKV)
// MODE 1: f32  out = acc + bias + resid             (proj, fc2)
// MODE 2: bf16 out = gelu(acc + bias)               (fc1)
template <int MODE>
__global__ __launch_bounds__(256)
void gemm_wmma_bf16(const __bf16* __restrict__ A, const __bf16* __restrict__ W,
                    const float* __restrict__ bias, const float* __restrict__ resid,
                    __bf16* __restrict__ out_bf, float* __restrict__ out_f,
                    int M, int N, int Kd) {
    const int lane   = threadIdx.x & 31;
    const int wave   = threadIdx.x >> 5;             // 8 waves per block
    const int m_tile = blockIdx.y * 256 + wave * 32; // 32 rows per wave
    const int n_base = blockIdx.x * 64;

    int arow0 = m_tile + (lane & 15);
    int arow1 = m_tile + 16 + (lane & 15);
    if (arow0 >= M) arow0 = M - 1;                   // clamp; stores guarded
    if (arow1 >= M) arow1 = M - 1;
    const int akh = (lane >> 4) << 3;                // 0 or 8
    const int bkh = (lane >> 4) << 3;                // 0 or 8
    const __bf16* aptr0 = A + (size_t)arow0 * Kd + akh;
    const __bf16* aptr1 = A + (size_t)arow1 * Kd + akh;
    const __bf16* bptr  = W + (size_t)(n_base + (lane & 15)) * Kd + bkh;
    const size_t  bstride = (size_t)16 * Kd;         // next 16-col tile

    v8f acc00 = {}, acc01 = {}, acc02 = {}, acc03 = {};
    v8f acc10 = {}, acc11 = {}, acc12 = {}, acc13 = {};
    for (int k0 = 0; k0 < Kd; k0 += 32) {
        __builtin_prefetch(aptr0 + k0 + 128, 0, 1);  // global_prefetch_b8
        v16bf a0 = load_a_frag(aptr0 + k0);
        v16bf a1 = load_a_frag(aptr1 + k0);
        v16bf b0 = load_b_frag(bptr + k0);
        v16bf b1 = load_b_frag(bptr + bstride + k0);
        v16bf b2 = load_b_frag(bptr + 2 * bstride + k0);
        v16bf b3 = load_b_frag(bptr + 3 * bstride + k0);
        acc00 = wmma_bf16(a0, b0, acc00);
        acc10 = wmma_bf16(a1, b0, acc10);
        acc01 = wmma_bf16(a0, b1, acc01);
        acc11 = wmma_bf16(a1, b1, acc11);
        acc02 = wmma_bf16(a0, b2, acc02);
        acc12 = wmma_bf16(a1, b2, acc12);
        acc03 = wmma_bf16(a0, b3, acc03);
        acc13 = wmma_bf16(a1, b3, acc13);
    }

    // C/D layout: lane -> N = n_base + t*16 + (lane&15); VGPR r -> M = r + 8*(lane>>4)
    const int mrow = m_tile + ((lane >> 4) << 3);
    const int ncol = lane & 15;
#pragma unroll
    for (int s = 0; s < 2; ++s) {
#pragma unroll
        for (int t = 0; t < 4; ++t) {
            v8f acc;
            if (s == 0)
                acc = (t == 0) ? acc00 : (t == 1) ? acc01 : (t == 2) ? acc02 : acc03;
            else
                acc = (t == 0) ? acc10 : (t == 1) ? acc11 : (t == 2) ? acc12 : acc13;
            const int n = n_base + t * 16 + ncol;
            const float bv = bias[n];
#pragma unroll
            for (int r = 0; r < 8; ++r) {
                const int m = mrow + s * 16 + r;
                if (m >= M) continue;
                float v = acc[r] + bv;
                const size_t idx = (size_t)m * N + n;
                if (MODE == 0) {
                    out_bf[idx] = f2bf(v);
                } else if (MODE == 1) {
                    out_f[idx] = v + resid[idx];
                } else {
                    v = 0.5f * v * (1.0f + erff(v * 0.70710678118654752f)); // exact gelu
                    out_bf[idx] = f2bf(v);
                }
            }
        }
    }
}

// -------------------------------------------------------------- attention
// one block per (n,d); one wave per head; lane owns dims {2*lane, 2*lane+1}
__global__ __launch_bounds__(256)
void attn_kernel(const __bf16* __restrict__ qkv, const int* __restrict__ which,
                 const int* __restrict__ mask, __bf16* __restrict__ y) {
    const int nd   = blockIdx.x;            // token index n*DEPTH+d
    const int n    = nd / DEPTH;
    const int d    = nd % DEPTH;
    const int h    = threadIdx.x >> 5;      // head
    const int lane = threadIdx.x & 31;
    const int c0   = lane * 2;

    const __bf16* qrow = qkv + (size_t)nd * NQKV + h * HD;
    const float q0 = bf2f(qrow[c0])     * 8.0f;   // * hd^0.5
    const float q1 = bf2f(qrow[c0 + 1]) * 8.0f;

    float sc[KNBR];
#pragma unroll
    for (int k = 0; k < KNBR; ++k) {
        const int wv = which[n * KNBR + k];
        const __bf16* krow =
            qkv + (size_t)(wv * DEPTH + d) * NQKV + CDIM + h * HD;
        float p = q0 * bf2f(krow[c0]) + q1 * bf2f(krow[c0 + 1]);
#pragma unroll
        for (int off = 16; off >= 1; off >>= 1) p += __shfl_xor(p, off, 32);
        sc[k] = (mask[n * KNBR + k] > 0) ? p : -1.0e9f;
    }
    float mx = sc[0];
#pragma unroll
    for (int k = 1; k < KNBR; ++k) mx = fmaxf(mx, sc[k]);
    float ssum = 0.0f;
#pragma unroll
    for (int k = 0; k < KNBR; ++k) { sc[k] = __expf(sc[k] - mx); ssum += sc[k]; }
    const float inv = 1.0f / ssum;

    float y0 = 0.0f, y1 = 0.0f;
#pragma unroll
    for (int k = 0; k < KNBR; ++k) {
        const int wv = which[n * KNBR + k];
        const __bf16* vrow =
            qkv + (size_t)(wv * DEPTH + d) * NQKV + 2 * CDIM + h * HD;
        y0 += sc[k] * bf2f(vrow[c0]);
        y1 += sc[k] * bf2f(vrow[c0 + 1]);
    }
    __bf16* yrow = y + (size_t)nd * CDIM + h * HD;
    yrow[c0]     = f2bf(y0 * inv);
    yrow[c0 + 1] = f2bf(y1 * inv);
}

// ------------------------------------------------------------------- host
extern "C" void kernel_launch(void* const* d_in, const int* in_sizes, int n_in,
                              void* d_out, int out_size, void* d_ws, size_t ws_size,
                              hipStream_t stream) {
    const float* x      = (const float*)d_in[0];
    const int*   which  = (const int*)  d_in[1];
    const int*   mask   = (const int*)  d_in[2];
    const float* ln1_w  = (const float*)d_in[3];
    const float* ln1_b  = (const float*)d_in[4];
    const float* qkv_w  = (const float*)d_in[5];
    const float* qkv_b  = (const float*)d_in[6];
    const float* proj_w = (const float*)d_in[7];
    const float* proj_b = (const float*)d_in[8];
    const float* ln2_w  = (const float*)d_in[9];
    const float* ln2_b  = (const float*)d_in[10];
    const float* fc1_w  = (const float*)d_in[11];
    const float* fc1_b  = (const float*)d_in[12];
    const float* fc2_w  = (const float*)d_in[13];
    const float* fc2_b  = (const float*)d_in[14];
    float* out = (float*)d_out;

    // ------------------------------------------------- workspace layout
    char* base = (char*)d_ws;
    size_t off = 0;
    auto alloc = [&](size_t bytes) -> char* {
        char* p = base + off;
        off = (off + bytes + 255) & ~(size_t)255;
        return p;
    };
    __bf16* Wqkv  = (__bf16*)alloc((size_t)NQKV * CDIM * 2);
    __bf16* Wproj = (__bf16*)alloc((size_t)CDIM * CDIM * 2);
    __bf16* Wfc1  = (__bf16*)alloc((size_t)HID * CDIM * 2);
    __bf16* Wfc2  = (__bf16*)alloc((size_t)CDIM * HID * 2);
    __bf16* xn    = (__bf16*)alloc((size_t)MROWS * CDIM * 2);
    __bf16* qkvh  = (__bf16*)alloc((size_t)MROWS * HID * 2);  // qkv, reused as fc1 act
    __bf16* ybuf  = (__bf16*)alloc((size_t)MROWS * CDIM * 2);
    float*  x2    = (float*) alloc((size_t)MROWS * CDIM * 4);
    __bf16* xn2   = (__bf16*)alloc((size_t)MROWS * CDIM * 2);
    (void)ws_size; (void)n_in; (void)in_sizes; (void)out_size;

    // 1) weights -> bf16
    cast_f32_to_bf16<<<1024, 256, 0, stream>>>(qkv_w,  Wqkv,  NQKV * CDIM);
    cast_f32_to_bf16<<<1024, 256, 0, stream>>>(proj_w, Wproj, CDIM * CDIM);
    cast_f32_to_bf16<<<1024, 256, 0, stream>>>(fc1_w,  Wfc1,  HID * CDIM);
    cast_f32_to_bf16<<<1024, 256, 0, stream>>>(fc2_w,  Wfc2,  CDIM * HID);

    const dim3 blk(256);
    const unsigned gy = (MROWS + 255) / 256;   // 256 rows per block (8 waves x 32)

    // 2) LN1
    layernorm_bf16<<<MROWS, blk, 0, stream>>>(x, ln1_w, ln1_b, xn);

    // 3) QKV GEMM: [M,1536] = xn[M,512] @ Wqkv^T
    gemm_wmma_bf16<0><<<dim3(NQKV / 64, gy), blk, 0, stream>>>(
        xn, Wqkv, qkv_b, nullptr, qkvh, nullptr, MROWS, NQKV, CDIM);

    // 4) neighbor attention
    attn_kernel<<<MROWS, blk, 0, stream>>>(qkvh, which, mask, ybuf);

    // 5) proj GEMM + residual: x2 = x + y @ Wproj^T + b
    gemm_wmma_bf16<1><<<dim3(CDIM / 64, gy), blk, 0, stream>>>(
        ybuf, Wproj, proj_b, x, nullptr, x2, MROWS, CDIM, CDIM);

    // 6) LN2
    layernorm_bf16<<<MROWS, blk, 0, stream>>>(x2, ln2_w, ln2_b, xn2);

    // 7) FC1 + exact GELU (qkv buffer is dead -> reuse as h)
    gemm_wmma_bf16<2><<<dim3(HID / 64, gy), blk, 0, stream>>>(
        xn2, Wfc1, fc1_b, nullptr, qkvh, nullptr, MROWS, HID, CDIM);

    // 8) FC2 + residual -> output
    gemm_wmma_bf16<1><<<dim3(CDIM / 64, gy), blk, 0, stream>>>(
        qkvh, Wfc2, fc2_b, x2, nullptr, out, MROWS, CDIM, HID);
}
